// FloatingRegionScore_64304250356198
// MI455X (gfx1250) — compile-verified, matrix operand order checked by reading, not run on hardware
//
#include <hip/hip_runtime.h>

// CDNA5 / gfx1250, wave32. RIPU score:
//   score = region_impurity * (box33(entropy)/count)
// 33x33 box filters as WMMA matmuls against constant banded-ones Toeplitz
// f16 operands (exact for the one-hot histogram, f32 accumulate). Halo tiles
// staged LDS-side by the Tensor Data Mover (tensor_load_to_lds + TENSORcnt).

typedef __attribute__((ext_vector_type(16))) _Float16 v16h;
typedef __attribute__((ext_vector_type(8)))  _Float16 v8h;
typedef __attribute__((ext_vector_type(4)))  _Float16 v4h;
typedef __attribute__((ext_vector_type(8)))  float    v8f;
typedef __attribute__((ext_vector_type(4)))  unsigned int v4u;
typedef __attribute__((ext_vector_type(4)))  int      v4i;
typedef __attribute__((ext_vector_type(8)))  int      v8i;

#define NC       19
#define H        1024
#define W        2048
#define HW       (H * W)
#define HP       (H + 32)      // halo-padded intermediate images
#define WP       (W + 32)
#define EPS      1e-6f
#define INV_LOGC 0.33962327f   // 1 / ln(19)

// Half-index -> K mapping for the 16-bit A-operand VGPR layout
// (ISA 7.12.2: VGPR j, lane halves select K groups).
__device__ __forceinline__ int a_k(int i, int g) {
  int j = i >> 1, t = i & 1;
  return ((j & 4) << 2) + ((j & 3) << 1) + (g << 3) + t;
}

// ---------------------------------------------------------------------------
// Per-pixel softmax entropy + argmax from 19 logits held in registers.
// ---------------------------------------------------------------------------
__device__ __forceinline__ void softmax_ent(const float* v, float& hOut, int& amOut) {
  float m = v[0]; int am = 0;
  #pragma unroll
  for (int c = 1; c < NC; ++c) {
    if (v[c] > m) { m = v[c]; am = c; }
  }
  float e[NC], s = 0.f;
  #pragma unroll
  for (int c = 0; c < NC; ++c) { e[c] = __expf(v[c] - m); s += e[c]; }
  float inv = 1.f / s;
  float h = 0.f;
  #pragma unroll
  for (int c = 0; c < NC; ++c) {
    float pp = e[c] * inv;
    h -= pp * __logf(pp + EPS);
  }
  hOut = h * INV_LOGC;
  amOut = am;
}

// ---------------------------------------------------------------------------
// Kernel 1: entropy (f16) + argmax (u8) into halo-padded images (border:
// ent=0, pred=255 => one-hot 0) so kernel 2's TDM tiles never go OOB.
// 4 pixels/thread: 19x global_load_b128 -> this is the 159 MB roofline pass.
// ---------------------------------------------------------------------------
__global__ __launch_bounds__(256) void ent_argmax_kernel(
    const float* __restrict__ logit,
    _Float16* __restrict__ entP,
    unsigned char* __restrict__ predP) {
  int quad = blockIdx.x * blockDim.x + threadIdx.x;   // one quad = 4 px
  int idx  = quad * 4;
  if (idx >= HP * WP) return;
  int y  = idx / WP, x = idx - y * WP;                // WP%4==0: same row
  int gy = y - 16,   gx = x - 16;

  const bool rowIn  = (unsigned)gy < (unsigned)H;
  const bool allIn  = rowIn && (gx >= 0) && (gx + 3 < W);

  if (allIn) {
    // ---- fast path: 19 x b128 loads, 4 pixels computed sequentially ----
    const size_t p = (size_t)gy * W + gx;             // 16B aligned
    float vv[NC][4];
    #pragma unroll
    for (int c = 0; c < NC; ++c) {
      float4 t = *(const float4*)(logit + (size_t)c * HW + p);
      vv[c][0] = t.x; vv[c][1] = t.y; vv[c][2] = t.z; vv[c][3] = t.w;
    }
    v4h eo;
    unsigned po = 0;
    #pragma unroll
    for (int u = 0; u < 4; ++u) {
      float vcol[NC];
      #pragma unroll
      for (int c = 0; c < NC; ++c) vcol[c] = vv[c][u];
      float h; int am;
      softmax_ent(vcol, h, am);
      eo[u] = (_Float16)h;
      po |= ((unsigned)am & 0xffu) << (8 * u);
    }
    *(v4h*)&entP[idx] = eo;                 // 8B store
    *(unsigned*)&predP[idx] = po;           // 4B store
  } else {
    // ---- border path: per-pixel scalar ----
    #pragma unroll
    for (int u = 0; u < 4; ++u) {
      int gxu = gx + u;
      if (rowIn && (unsigned)gxu < (unsigned)W) {
        const size_t p = (size_t)gy * W + gxu;
        float vcol[NC];
        #pragma unroll
        for (int c = 0; c < NC; ++c) vcol[c] = logit[(size_t)c * HW + p];
        float h; int am;
        softmax_ent(vcol, h, am);
        entP[idx + u]  = (_Float16)h;
        predP[idx + u] = (unsigned char)am;
      } else {
        entP[idx + u]  = (_Float16)0.0f;
        predP[idx + u] = (unsigned char)255;   // no class -> zero padding
      }
    }
  }
}

// ---------------------------------------------------------------------------
// TDM: 2D tile load Global -> LDS via tensor descriptor (D# groups per ISA
// CDNA5 ISA). LDS row padding done by the pad engine.
// ---------------------------------------------------------------------------
__device__ __forceinline__ unsigned lds_off(const void* p) {
  return (unsigned)(unsigned long long)(uintptr_t)p;   // addr[31:0] = LDS offset
}

__device__ __forceinline__ void tdm_load_2d(
    unsigned ldsAddr, const void* gtile,
    unsigned dataSizeLog,                 // 0 = 1B, 1 = 2B elements
    unsigned tensorW, unsigned tensorH,   // full (padded-image) dims, elements
    unsigned long long rowStride,         // elements between rows
    unsigned tileW, unsigned tileH,       // tile dims, elements
    unsigned padInterval, unsigned padAmount) {  // D# pad codes
  unsigned long long ga = (unsigned long long)(uintptr_t)gtile;

  v4u g0;
  g0[0] = 1u;                                           // count=1, user mode
  g0[1] = ldsAddr;                                      // lds_addr
  g0[2] = (unsigned)(ga & 0xffffffffull);               // global_addr[31:0]
  g0[3] = (unsigned)((ga >> 32) & 0x1ffffffull)         // global_addr[56:32]
        | (2u << 30);                                   // type = 2 ("image")

  v8i g1;
  g1[0] = (int)((dataSizeLog << 16) | (1u << 20)        // data_size, pad_enable
              | (padInterval << 22) | (padAmount << 25));
  g1[1] = (int)((tensorW & 0xffffu) << 16);             // tensor_dim0[15:0]
  g1[2] = (int)(((tensorW >> 16) & 0xffffu)             // tensor_dim0[31:16]
              | ((tensorH & 0xffffu) << 16));           // tensor_dim1[15:0]
  g1[3] = (int)(((tensorH >> 16) & 0xffffu)             // tensor_dim1[31:16]
              | ((tileW & 0xffffu) << 16));             // tile_dim0
  g1[4] = (int)(tileH & 0xffffu);                       // tile_dim1 (tile_dim2=0)
  g1[5] = (int)(rowStride & 0xffffffffull);             // tensor_dim0_stride[31:0]
  g1[6] = (int)((rowStride >> 32) & 0xffffull);         // stride[47:32]
  g1[7] = 0;

  v4i z = {};
#if defined(__clang_major__) && (__clang_major__ >= 23)
  v8i z8 = {};
  __builtin_amdgcn_tensor_load_to_lds(g0, g1, z, z, z8, 0);
#else
  __builtin_amdgcn_tensor_load_to_lds(g0, g1, z, z, 0);
#endif
}

// ---------------------------------------------------------------------------
// Kernel 2: fused separable 33x33 box sums (WMMA) + impurity/uncertainty.
// 128 threads / 4 waves own a 32x32 output tile; one wave = one 16x16 WMMA
// tile. 64x64 halo of pred/ent staged by TDM into LDS.
// ---------------------------------------------------------------------------
__global__ __launch_bounds__(128) void ripu_box_kernel(
    const _Float16* __restrict__ entP,
    const unsigned char* __restrict__ predP,
    float* __restrict__ outScore,
    float* __restrict__ outImp,
    float* __restrict__ outPU) {

  __shared__ unsigned char predT[64][72];   // halo classes (TDM pad: 2 dw/16)
  __shared__ _Float16      entT[64][72];    // halo entropy (TDM pad: 4 dw/32)
  __shared__ _Float16      rowSt[32][72];   // row-pass result, TRANSPOSED:
                                            // [output col][band row]

  const int tid  = threadIdx.x;
  const int lane = tid & 31;
  const int wave = tid >> 5;
  const int m    = lane & 15;   // A-operand row / C,D column
  const int g    = lane >> 4;   // lane-half selector

  const int R  = blockIdx.y * 32;   // output row base
  const int Cc = blockIdx.x * 32;   // output col base

  // ---- TDM-stage 64x64 halo tiles (padded-image coords start at R,Cc) ----
  if (wave == 0) {
    tdm_load_2d(lds_off(&predT[0][0]),
                predP + (size_t)R * WP + Cc,
                0, WP, HP, (unsigned long long)WP, 64, 64,
                /*interval 16dw=64B*/3, /*amount 2dw=8B*/1);
  } else if (wave == 1) {
    tdm_load_2d(lds_off(&entT[0][0]),
                entP + (size_t)R * WP + Cc,
                1, WP, HP, (unsigned long long)WP, 64, 64,
                /*interval 32dw=128B*/4, /*amount 4dw=16B*/3);
  }
  __builtin_amdgcn_s_wait_tensorcnt(0);
  __syncthreads();

  // ---- constant banded-ones Toeplitz operands (K=32 chunk, K=16 chunk) ----
  v16h tLo, tHi;
  #pragma unroll
  for (int i = 0; i < 16; ++i) {
    int k = a_k(i, g);
    tLo[i] = (k >= m)           ? (_Float16)1.0f : (_Float16)0.0f;
    tHi[i] = (k <= m && k < 16) ? (_Float16)1.0f : (_Float16)0.0f;
  }

  // ---- preload row-pass A-source values once (reused for all channels) ----
  const int br = 16 * wave + m;           // LDS band row this lane feeds
  unsigned char pA1[2][16], pA2[2][8];
  _Float16      eA1[2][16], eA2[2][8];
  #pragma unroll
  for (int q = 0; q < 2; ++q) {
    #pragma unroll
    for (int i = 0; i < 16; ++i) {
      int k = a_k(i, g);
      pA1[q][i] = predT[br][16 * q + k];
      eA1[q][i] = entT[br][16 * q + k];
    }
    #pragma unroll
    for (int i = 0; i < 8; ++i) {
      int k = a_k(i, g);
      pA2[q][i] = predT[br][16 * q + 32 + k];
      eA2[q][i] = entT[br][16 * q + 32 + k];
    }
  }

  // ---- analytic window population count (box sum of ones) ----
  const int wr = wave >> 1, wq = wave & 1;
  const int ocol = Cc + 16 * wq + m;
  const float cntX = (float)(min(ocol + 16, W - 1) - max(ocol - 16, 0) + 1);
  float invCnt[8];
  #pragma unroll
  for (int r = 0; r < 8; ++r) {
    int orow = R + 16 * wr + r + 8 * g;
    float cntY = (float)(min(orow + 16, H - 1) - max(orow - 16, 0) + 1);
    invCnt[r] = 1.0f / (cntX * cntY);
  }

  v8f impAcc = {};   // sum_c -d*log(d+eps)
  v8f entBox = {};   // 33x33 box sum of normalized entropy

  for (int c = 0; c <= NC; ++c) {   // c == NC -> entropy plane
    // -- build row-pass A operands (one-hot of pred, or entropy values) --
    v16h A1[2], A2[2];
    if (c < NC) {
      int cc = c;
      #pragma unroll
      for (int q = 0; q < 2; ++q) {
        #pragma unroll
        for (int i = 0; i < 16; ++i)
          A1[q][i] = (pA1[q][i] == cc) ? (_Float16)1.0f : (_Float16)0.0f;
        #pragma unroll
        for (int i = 0; i < 8; ++i)
          A2[q][i] = (pA2[q][i] == cc) ? (_Float16)1.0f : (_Float16)0.0f;
        #pragma unroll
        for (int i = 8; i < 16; ++i) A2[q][i] = (_Float16)0.0f;
      }
    } else {
      #pragma unroll
      for (int q = 0; q < 2; ++q) {
        #pragma unroll
        for (int i = 0; i < 16; ++i) A1[q][i] = eA1[q][i];
        #pragma unroll
        for (int i = 0; i < 8; ++i)  A2[q][i] = eA2[q][i];
        #pragma unroll
        for (int i = 8; i < 16; ++i) A2[q][i] = (_Float16)0.0f;
      }
    }

    // -- row pass: horizontal 33-tap box sum = 2 WMMAs per 16x16 tile;
    //    store transposed, one ds_store_b128 per lane per q --
    #pragma unroll
    for (int q = 0; q < 2; ++q) {
      v8f rs = {};
      rs = __builtin_amdgcn_wmma_f32_16x16x32_f16(false, A1[q], false, tLo,
                                                  (short)0, rs, false, false);
      rs = __builtin_amdgcn_wmma_f32_16x16x32_f16(false, A2[q], false, tHi,
                                                  (short)0, rs, false, false);
      v8h packed;
      #pragma unroll
      for (int r = 0; r < 8; ++r) packed[r] = (_Float16)rs[r];
      *(v8h*)&rowSt[16 * q + m][16 * wave + 8 * g] = packed;   // 16B aligned
    }
    __syncthreads();

    // -- column pass: vertical 33-tap box sum = 2 WMMAs; B operands are
    //    three aligned ds_load_b128 per lane from the transposed buffer --
    const _Float16* colBase = &rowSt[16 * wq + m][0];
    v8h lo  = *(const v8h*)(colBase + 16 * wr + 8 * g);
    v8h hi  = *(const v8h*)(colBase + 16 * wr + 16 + 8 * g);
    v8h b2l = *(const v8h*)(colBase + 16 * wr + 32 + 8 * g);
    v8h zz  = {};
    v16h B1 = __builtin_shufflevector(lo, hi, 0, 1, 2, 3, 4, 5, 6, 7,
                                      8, 9, 10, 11, 12, 13, 14, 15);
    v16h B2 = __builtin_shufflevector(b2l, zz, 0, 1, 2, 3, 4, 5, 6, 7,
                                      8, 9, 10, 11, 12, 13, 14, 15);

    v8f s = {};
    s = __builtin_amdgcn_wmma_f32_16x16x32_f16(false, tLo, false, B1,
                                               (short)0, s, false, false);
    s = __builtin_amdgcn_wmma_f32_16x16x32_f16(false, tHi, false, B2,
                                               (short)0, s, false, false);

    if (c < NC) {
      #pragma unroll
      for (int r = 0; r < 8; ++r) {
        float d = s[r] * invCnt[r];          // local class fraction
        impAcc[r] -= d * __logf(d + EPS);
      }
    } else {
      entBox = s;
    }
    __syncthreads();   // rowSt free for next channel
  }

  // ---- finalize: score, impurity, prediction uncertainty ----
  #pragma unroll
  for (int r = 0; r < 8; ++r) {
    int orow = R + 16 * wr + r + 8 * g;
    int o = orow * W + ocol;
    float imp = impAcc[r] * INV_LOGC;
    float pu  = entBox[r] * invCnt[r];
    outScore[o] = imp * pu;
    outImp[o]   = imp;
    outPU[o]    = pu;
  }
}

// ---------------------------------------------------------------------------
extern "C" void kernel_launch(void* const* d_in, const int* in_sizes, int n_in,
                              void* d_out, int out_size, void* d_ws, size_t ws_size,
                              hipStream_t stream) {
  const float* logit = (const float*)d_in[0];
  float* out = (float*)d_out;

  // workspace: halo-padded ent f16 [HP*WP] (~4.4 MB) | pred u8 [HP*WP] (~2.2 MB)
  const size_t padN = (size_t)HP * WP;
  _Float16* entP = (_Float16*)d_ws;
  unsigned char* predP = (unsigned char*)d_ws + padN * sizeof(_Float16);

  ent_argmax_kernel<<<(HP * WP / 4 + 255) / 256, 256, 0, stream>>>(logit, entP, predP);

  dim3 grid(W / 32, H / 32);
  ripu_box_kernel<<<grid, 128, 0, stream>>>(entP, predP,
                                            out, out + HW, out + 2 * (size_t)HW);
}
